// Model_35502199668858
// MI455X (gfx1250) — matrix operand order repaired and checked by reading, not compiled, not run on hardware
//
#include <hip/hip_runtime.h>

typedef __attribute__((ext_vector_type(16))) _Float16 v16h;
typedef __attribute__((ext_vector_type(8)))  float    v8f;

#define BATCH 16
#define NPTS  4096
#define KS    64
#define EPSV  1e-5f

// ---------------------------------------------------------------------------
// Elementwise: (B,3,N) -> (B,N,3)
// ---------------------------------------------------------------------------
__global__ void xpose_kernel(const float* __restrict__ pc, float* __restrict__ xyz, int n)
{
    long long e = (long long)blockIdx.x * blockDim.x + threadIdx.x;
    long long total = (long long)BATCH * n * 3;
    if (e >= total) return;
    int c = (int)(e % 3);
    int i = (int)((e / 3) % n);
    int b = (int)(e / (3LL * n));
    xyz[e] = pc[((size_t)b * 3 + c) * n + i];
}

// ---------------------------------------------------------------------------
// Farthest point sampling: one block per batch, sequential scan with
// block-wide argmax reduction each step. Matches CUDA convention (start at 0).
// ---------------------------------------------------------------------------
__global__ void fps_kernel(const float* __restrict__ xyz, // [B][N][3]
                           int* __restrict__ out,         // [B][npoint]
                           int N, int npoint)
{
    __shared__ float dist[NPTS];
    __shared__ float rv[256];
    __shared__ int   ri[256];
    __shared__ float lastxyz[3];
    const int b = blockIdx.x, t = threadIdx.x, T = blockDim.x;
    const float* base = xyz + (size_t)b * N * 3;
    for (int i = t; i < N; i += T) dist[i] = 1e10f;
    if (t == 0) {
        out[(size_t)b * npoint] = 0;
        lastxyz[0] = base[0]; lastxyz[1] = base[1]; lastxyz[2] = base[2];
    }
    __syncthreads();
    for (int j = 1; j < npoint; ++j) {
        const float lx = lastxyz[0], ly = lastxyz[1], lz = lastxyz[2];
        float bv = -1.0f; int bi = 0;
        for (int i = t; i < N; i += T) {
            float dx = base[i*3+0] - lx;
            float dy = base[i*3+1] - ly;
            float dz = base[i*3+2] - lz;
            float d  = dx*dx + dy*dy + dz*dz;
            d = fminf(dist[i], d);
            dist[i] = d;
            if (d > bv) { bv = d; bi = i; }
        }
        rv[t] = bv; ri[t] = bi;
        __syncthreads();
        for (int s = T >> 1; s > 0; s >>= 1) {
            if (t < s && rv[t + s] > rv[t]) { rv[t] = rv[t + s]; ri[t] = ri[t + s]; }
            __syncthreads();
        }
        if (t == 0) {
            int nxt = ri[0];
            out[(size_t)b * npoint + j] = nxt;
            lastxyz[0] = base[nxt*3+0];
            lastxyz[1] = base[nxt*3+1];
            lastxyz[2] = base[nxt*3+2];
        }
        __syncthreads();
    }
}

__global__ void gather_xyz_kernel(const float* __restrict__ xyz, const int* __restrict__ fidx,
                                  float* __restrict__ nx, int N, int S)
{
    long long e = (long long)blockIdx.x * blockDim.x + threadIdx.x;
    long long total = (long long)BATCH * S * 3;
    if (e >= total) return;
    int c = (int)(e % 3);
    int s = (int)((e / 3) % S);
    int b = (int)(e / (3LL * S));
    nx[e] = xyz[((size_t)b * N + fidx[(size_t)b * S + s]) * 3 + c];
}

// ---------------------------------------------------------------------------
// Ball query: one wave32 per center; ordered first-K selection via ballot +
// prefix popcount so index order matches jnp.sort semantics.
// ---------------------------------------------------------------------------
__global__ void ball_query_kernel(const float* __restrict__ xyz,  // [B][N][3]
                                  const float* __restrict__ ctr,  // [B][S][3]
                                  int* __restrict__ idx,          // [B][S][K]
                                  int N, int S, float r2)
{
    int gw = (int)(((long long)blockIdx.x * blockDim.x + threadIdx.x) >> 5);
    int lane = threadIdx.x & 31;
    if (gw >= BATCH * S) return;
    int b = gw / S, s = gw % S;
    const float* base = xyz + (size_t)b * N * 3;
    const float cx = ctr[((size_t)b * S + s) * 3 + 0];
    const float cy = ctr[((size_t)b * S + s) * 3 + 1];
    const float cz = ctr[((size_t)b * S + s) * 3 + 2];
    int* o = idx + ((size_t)b * S + s) * KS;
    int cnt = 0, first = 0;
    for (int i0 = 0; i0 < N && cnt < KS; i0 += 32) {
        int i = i0 + lane;
        bool hit = false;
        if (i < N) {
            float dx = base[i*3+0] - cx;
            float dy = base[i*3+1] - cy;
            float dz = base[i*3+2] - cz;
            hit = (dx*dx + dy*dy + dz*dz) < r2;
        }
        unsigned mask = (unsigned)__ballot(hit);
        if (cnt == 0 && mask) first = i0 + __ffs(mask) - 1;
        int pre = __popc(mask & ((1u << lane) - 1u));
        if (hit && (cnt + pre) < KS) o[cnt + pre] = i;
        cnt += __popc(mask);
    }
    if (cnt > KS) cnt = KS;
    int pad = (cnt == 0) ? 0 : first;
    for (int k = cnt + lane; k < KS; k += 32) o[k] = pad;
}

// ---------------------------------------------------------------------------
// Grouping -> f16 GEMM operand X [P][Cpad], P = B*S*K.
// channels 0..2 = centered xyz, 3..3+Cf-1 = gathered feats, rest zero pad.
// ---------------------------------------------------------------------------
__global__ void group_kernel(const float* __restrict__ xyz,   // [B][N][3]
                             const float* __restrict__ ctr,   // [B][S][3]
                             const float* __restrict__ feats, // [B][Cf][N] or null
                             const int* __restrict__ idx,     // [B][S][K]
                             _Float16* __restrict__ Xh,       // [B*S*K][Cpad]
                             int N, int S, int Cf, int Cpad)
{
    long long e = (long long)blockIdx.x * blockDim.x + threadIdx.x;
    long long total = (long long)BATCH * S * KS * Cpad;
    if (e >= total) return;
    int c = (int)(e % Cpad);
    long long p = e / Cpad;
    int k = (int)(p % KS);
    long long bs = p / KS;
    int s = (int)(bs % S);
    int b = (int)(bs / S);
    int id = idx[((size_t)b * S + s) * KS + k];
    float v = 0.0f;
    if (c < 3)
        v = xyz[((size_t)b * N + id) * 3 + c] - ctr[((size_t)b * S + s) * 3 + c];
    else if (c < 3 + Cf)
        v = feats[((size_t)b * Cf + (c - 3)) * N + id];
    Xh[e] = (_Float16)v;
}

// group-all for SA3: concat(xyz(B,S,3) as channels 0..2, feats(B,Cf,S))
__global__ void groupall_kernel(const float* __restrict__ nx,    // [B][S][3]
                                const float* __restrict__ feats, // [B][Cf][S]
                                _Float16* __restrict__ Xh,       // [B*S][Cpad]
                                int S, int Cf, int Cpad)
{
    long long e = (long long)blockIdx.x * blockDim.x + threadIdx.x;
    long long total = (long long)BATCH * S * Cpad;
    if (e >= total) return;
    int c = (int)(e % Cpad);
    int s = (int)((e / Cpad) % S);
    int b = (int)(e / ((long long)Cpad * S));
    float v = 0.0f;
    if (c < 3)            v = nx[((size_t)b * S + s) * 3 + c];
    else if (c < 3 + Cf)  v = feats[((size_t)b * Cf + (c - 3)) * S + s];
    Xh[e] = (_Float16)v;
}

// ---------------------------------------------------------------------------
// Weights f32 (O,Cin) -> f16 pre-swizzled into this GEMM's A-fragment order (contiguous k):
// layout [mt][ks][lane][16], so the GEMM reads one contiguous v16h per lane
// per K-step (fully coalesced 1KB per wave per step, zero repack VALU).
//   lane row m = mt*16 + (lane&15), half = lane>>4
//   elem e -> K = ks*32 + half*16 + e
// ---------------------------------------------------------------------------
__global__ void wcvt_swz_kernel(const float* __restrict__ W, _Float16* __restrict__ WhS,
                                int O, int Cin, int Cpad)
{
    const int ksteps = Cpad >> 5;
    long long id = (long long)blockIdx.x * blockDim.x + threadIdx.x;
    long long total = (long long)O * Cpad;
    if (id >= total) return;
    int e    = (int)(id & 15);
    int lane = (int)((id >> 4) & 31);
    long long rest = id >> 9;
    int ks = (int)(rest % ksteps);
    int mt = (int)(rest / ksteps);
    int half = lane >> 4, lm = lane & 15;
    int m = mt * 16 + lm;
    int K = ks * 32 + half * 16 + e;
    WhS[id] = (K < Cin) ? (_Float16)W[(size_t)m * Cin + K] : (_Float16)0.0f;
}

// ---------------------------------------------------------------------------
// WMMA GEMM: Y[P][O] = W(O x C) * X(P x C)^T, f16 in / f32 acc.
// One wave owns a 16(M) x 64(N) super-tile (A reused across 4 B tiles).
// P is always a multiple of 64 (callers pad), so there is no tail logic and
// every branch is scalar (wave id via readfirstlane) => EXEC stays all-ones
// for every v_wmma as the ISA requires.
// Software-pipelined K loop: fragments for step ks+1 are loaded before the
// 4 WMMAs of step ks issue, hiding global-load latency behind matrix math.
// Fragment layouts per CDNA5 ISA (wave32):
//   B 32x16 f16: lane col = lane&15; elems 0..15 -> K = half*16+e  (one v16h; k order chosen here to match A, not the ISA table)
//   D 16x16 f32: lane col = lane&15; elem r -> row = half*8+r      (one v8f)
// ---------------------------------------------------------------------------
__global__ void wmma_gemm_kernel(const _Float16* __restrict__ WhS, // swizzled [O/16][C/32][32][16]
                                 const _Float16* __restrict__ Xh,  // [P][Cpad]
                                 float* __restrict__ Y,            // [P][O]
                                 int O, int Cpad, int P)
{
    const int lane = threadIdx.x & 31;
    const int wave = __builtin_amdgcn_readfirstlane(
        (int)(((long long)blockIdx.x * blockDim.x + threadIdx.x) >> 5));
    const int mtiles  = O >> 4;
    const int ngroups = P >> 6;            // P is a multiple of 64
    if (wave >= mtiles * ngroups) return;  // scalar branch
    const int mt = wave % mtiles;
    const int ng = wave / mtiles;
    const int half = lane >> 4;
    const int lm = lane & 15;
    const int ksteps = Cpad >> 5;

    const _Float16* wl = WhS + (((size_t)mt * ksteps) * 32 + lane) * 16;
    const _Float16* x0 = Xh + (size_t)(ng * 64 + lm) * Cpad + half * 16;
    const size_t xs = (size_t)16 * Cpad;

    v16h a  = *(const v16h*)(wl);
    v16h b0 = *(const v16h*)(x0);
    v16h b1 = *(const v16h*)(x0 + xs);
    v16h b2 = *(const v16h*)(x0 + 2 * xs);
    v16h b3 = *(const v16h*)(x0 + 3 * xs);

    v8f acc0 = {}, acc1 = {}, acc2 = {}, acc3 = {};
    for (int ks = 0; ks < ksteps; ++ks) {
        v16h ca = a, cb0 = b0, cb1 = b1, cb2 = b2, cb3 = b3;
        if (ks + 1 < ksteps) {             // scalar branch
            const _Float16* xn = x0 + (ks + 1) * 32;
            a  = *(const v16h*)(wl + (size_t)(ks + 1) * 512);
            b0 = *(const v16h*)(xn);
            b1 = *(const v16h*)(xn + xs);
            b2 = *(const v16h*)(xn + 2 * xs);
            b3 = *(const v16h*)(xn + 3 * xs);
        }
        acc0 = __builtin_amdgcn_wmma_f32_16x16x32_f16(false, ca, false, cb0, (short)0, acc0, false, false);
        acc1 = __builtin_amdgcn_wmma_f32_16x16x32_f16(false, ca, false, cb1, (short)0, acc1, false, false);
        acc2 = __builtin_amdgcn_wmma_f32_16x16x32_f16(false, ca, false, cb2, (short)0, acc2, false, false);
        acc3 = __builtin_amdgcn_wmma_f32_16x16x32_f16(false, ca, false, cb3, (short)0, acc3, false, false);
    }
    float* yb = Y + (size_t)(ng * 64 + lm) * O + mt * 16 + half * 8;
    const size_t ys = (size_t)16 * O;
    *(v8f*)(yb)          = acc0;
    *(v8f*)(yb + ys)     = acc1;
    *(v8f*)(yb + 2 * ys) = acc2;
    *(v8f*)(yb + 3 * ys) = acc3;
}

// ---------------------------------------------------------------------------
// BN training statistics per output channel over first P rows of Y
// ---------------------------------------------------------------------------
__global__ void bn_stats_kernel(const float* __restrict__ Y, // [P][O]
                                const float* __restrict__ g,
                                const float* __restrict__ bt,
                                float* __restrict__ scale,
                                float* __restrict__ shift,
                                int P, int O)
{
    __shared__ float ss[256], sq[256];
    const int o = blockIdx.x, t = threadIdx.x, T = blockDim.x;
    float s1 = 0.0f, s2 = 0.0f;
    for (int p = t; p < P; p += T) {
        float v = Y[(size_t)p * O + o];
        s1 += v; s2 += v * v;
    }
    ss[t] = s1; sq[t] = s2;
    __syncthreads();
    for (int r = T >> 1; r > 0; r >>= 1) {
        if (t < r) { ss[t] += ss[t + r]; sq[t] += sq[t + r]; }
        __syncthreads();
    }
    if (t == 0) {
        float mean = ss[0] / (float)P;
        float var  = sq[0] / (float)P - mean * mean;
        float sc   = g[o] * rsqrtf(var + EPSV);
        scale[o] = sc;
        shift[o] = bt[o] - mean * sc;
    }
}

// normalize + relu -> f16 activations (rows >= Preal zero-padded)
__global__ void bn_relu_f16_kernel(const float* __restrict__ Y,
                                   const float* __restrict__ scale,
                                   const float* __restrict__ shift,
                                   _Float16* __restrict__ Xh,
                                   long long Ptot, long long Preal, int O)
{
    long long e = (long long)blockIdx.x * blockDim.x + threadIdx.x;
    long long total = Ptot * O;
    if (e >= total) return;
    int c = (int)(e % O);
    long long p = e / O;
    float v = 0.0f;
    if (p < Preal) v = fmaxf(Y[e] * scale[c] + shift[c], 0.0f);
    Xh[e] = (_Float16)v;
}

// normalize + relu + max over K -> feats f32 [B][O][S]
__global__ void bn_relu_max_kernel(const float* __restrict__ Y, // [(b*S+s)*K+k][O]
                                   const float* __restrict__ scale,
                                   const float* __restrict__ shift,
                                   float* __restrict__ feats,   // [B][O][S]
                                   int S, int O)
{
    long long e = (long long)blockIdx.x * blockDim.x + threadIdx.x;
    long long total = (long long)BATCH * S * O;
    if (e >= total) return;
    int o = (int)(e % O);
    int s = (int)((e / O) % S);
    int b = (int)(e / ((long long)O * S));
    const size_t base = ((size_t)b * S + s) * KS;
    const float sc = scale[o], sh = shift[o];
    float mx = 0.0f;
    for (int k = 0; k < KS; ++k) {
        float v = fmaxf(Y[(base + k) * O + o] * sc + sh, 0.0f);
        mx = fmaxf(mx, v);
    }
    feats[((size_t)b * O + o) * S + s] = mx;
}

// SA3 final: normalize + relu + max over S -> f16 [Bpad][O] (rows >= BATCH zero)
__global__ void bn_relu_max_all_kernel(const float* __restrict__ Y, // [(b*S+s)][O]
                                       const float* __restrict__ scale,
                                       const float* __restrict__ shift,
                                       _Float16* __restrict__ Xh,   // [Bpad][O]
                                       int S, int O, int Bpad)
{
    long long e = (long long)blockIdx.x * blockDim.x + threadIdx.x;
    long long total = (long long)Bpad * O;
    if (e >= total) return;
    int o = (int)(e % O);
    int b = (int)(e / O);
    float mx = 0.0f;
    if (b < BATCH) {
        const float sc = scale[o], sh = shift[o];
        for (int s = 0; s < S; ++s) {
            float v = fmaxf(Y[((size_t)b * S + s) * O + o] * sc + sh, 0.0f);
            mx = fmaxf(mx, v);
        }
    }
    Xh[(size_t)b * O + o] = (_Float16)mx;
}

__global__ void sigmoid_kernel(const float* __restrict__ Y, float* __restrict__ out, long long n)
{
    long long e = (long long)blockIdx.x * blockDim.x + threadIdx.x;
    if (e >= n) return;
    out[e] = 1.0f / (1.0f + expf(-Y[e]));
}

// ---------------------------------------------------------------------------
// Host helpers
// ---------------------------------------------------------------------------
static inline int cdiv(long long a, int b) { return (int)((a + b - 1) / b); }

static void launch_gemm(const _Float16* WhS, const _Float16* Xh, float* Y,
                        int O, int Cpad, int P, hipStream_t st)
{
    long long threads = (long long)(O / 16) * (P / 64) * 32;
    wmma_gemm_kernel<<<cdiv(threads, 256), 256, 0, st>>>(WhS, Xh, Y, O, Cpad, P);
}

// Pg: GEMM rows (multiple of 64); Ps: rows used for BN statistics
static void dense_bn(const float* W, const float* G, const float* Bb,
                     int O, int Cin, int Cpad, const _Float16* Xin, int Pg, int Ps,
                     float* Y, _Float16* WhS, float* scl, float* sft, hipStream_t st)
{
    long long we = (long long)O * Cpad;
    wcvt_swz_kernel<<<cdiv(we, 256), 256, 0, st>>>(W, WhS, O, Cin, Cpad);
    launch_gemm(WhS, Xin, Y, O, Cpad, Pg, st);
    bn_stats_kernel<<<O, 256, 0, st>>>(Y, G, Bb, scl, sft, Ps, O);
}

// ---------------------------------------------------------------------------
extern "C" void kernel_launch(void* const* d_in, const int* in_sizes, int n_in,
                              void* d_out, int out_size, void* d_ws, size_t ws_size,
                              hipStream_t stream)
{
    // ---- input mapping (handles insertion-order and jax sorted-key order) ----
    const float *Wl[9], *Gl[9], *Bl[9];
    const float *fcW1, *fcG1, *fcB1, *fcW2, *fcG2, *fcB2, *fcW3, *PC;
    if (n_in >= 35 && in_sizes[0] == BATCH * 3 * NPTS) {
        PC = (const float*)d_in[0];
        int t = 1;
        for (int l = 0; l < 9; ++l) {
            Wl[l] = (const float*)d_in[t++];
            Gl[l] = (const float*)d_in[t++];
            Bl[l] = (const float*)d_in[t++];
        }
        fcW1 = (const float*)d_in[t++]; fcG1 = (const float*)d_in[t++]; fcB1 = (const float*)d_in[t++];
        fcW2 = (const float*)d_in[t++]; fcG2 = (const float*)d_in[t++]; fcB2 = (const float*)d_in[t++];
        fcW3 = (const float*)d_in[t++];
    } else {
        fcB1 = (const float*)d_in[0]; fcB2 = (const float*)d_in[1];
        fcG1 = (const float*)d_in[2]; fcG2 = (const float*)d_in[3];
        fcW1 = (const float*)d_in[4]; fcW2 = (const float*)d_in[5];
        fcW3 = (const float*)d_in[6];
        int t = 7;
        for (int l = 0; l < 9; ++l) {
            Bl[l] = (const float*)d_in[t++];
            Gl[l] = (const float*)d_in[t++];
            Wl[l] = (const float*)d_in[t++];
        }
        PC = (const float*)d_in[34];
    }

    // ---- workspace bump allocator ----
    char* wsp = (char*)d_ws;
    size_t off = 0;
    auto alloc = [&](size_t bytes) -> void* {
        void* p = wsp + off;
        off += (bytes + 255) & ~(size_t)255;
        return p;
    };
    const size_t P1 = (size_t)BATCH * 512 * KS;   // 524288
    const size_t P2 = (size_t)BATCH * 128 * KS;   // 131072
    const size_t P3 = (size_t)BATCH * 128;        // 2048
    const int    PFC = 64;                        // FC rows padded 16 -> 64

    float*     xyz1   = (float*)alloc(sizeof(float) * BATCH * NPTS * 3);
    int*       fidx1  = (int*)alloc(sizeof(int) * BATCH * 512);
    float*     nx1    = (float*)alloc(sizeof(float) * BATCH * 512 * 3);
    int*       idx1   = (int*)alloc(sizeof(int) * BATCH * 512 * KS);
    int*       fidx2  = (int*)alloc(sizeof(int) * BATCH * 128);
    float*     nx2    = (float*)alloc(sizeof(float) * BATCH * 128 * 3);
    int*       idx2   = (int*)alloc(sizeof(int) * BATCH * 128 * KS);
    float*     feats1 = (float*)alloc(sizeof(float) * (size_t)BATCH * 128 * 512);
    float*     feats2 = (float*)alloc(sizeof(float) * (size_t)BATCH * 256 * 128);
    _Float16*  Xa     = (_Float16*)alloc(sizeof(_Float16) * P1 * 64);
    _Float16*  Xb     = (_Float16*)alloc(sizeof(_Float16) * P1 * 64);
    float*     Y      = (float*)alloc(sizeof(float) * P1 * 128);
    _Float16*  Wh     = (_Float16*)alloc(sizeof(_Float16) * 4096 * 256);
    float*     scl    = (float*)alloc(sizeof(float) * 4096);
    float*     sft    = (float*)alloc(sizeof(float) * 4096);
    if (off > ws_size) return; // insufficient scratch (deterministic)

    // ---- stage 0: transpose pointcloud ----
    xpose_kernel<<<cdiv((long long)BATCH * NPTS * 3, 256), 256, 0, stream>>>(PC, xyz1, NPTS);

    // ================= SA1 =================
    fps_kernel<<<BATCH, 256, 0, stream>>>(xyz1, fidx1, NPTS, 512);
    gather_xyz_kernel<<<cdiv((long long)BATCH * 512 * 3, 256), 256, 0, stream>>>(xyz1, fidx1, nx1, NPTS, 512);
    ball_query_kernel<<<cdiv((long long)BATCH * 512 * 32, 256), 256, 0, stream>>>(xyz1, nx1, idx1, NPTS, 512, 0.2f * 0.2f);
    group_kernel<<<cdiv((long long)P1 * 32, 256), 256, 0, stream>>>(xyz1, nx1, nullptr, idx1, Xa, NPTS, 512, 0, 32);
    // L0: 3 -> 64
    dense_bn(Wl[0], Gl[0], Bl[0], 64, 3, 32, Xa, (int)P1, (int)P1, Y, Wh, scl, sft, stream);
    bn_relu_f16_kernel<<<cdiv((long long)P1 * 64, 256), 256, 0, stream>>>(Y, scl, sft, Xb, (long long)P1, (long long)P1, 64);
    // L1: 64 -> 64
    dense_bn(Wl[1], Gl[1], Bl[1], 64, 64, 64, Xb, (int)P1, (int)P1, Y, Wh, scl, sft, stream);
    bn_relu_f16_kernel<<<cdiv((long long)P1 * 64, 256), 256, 0, stream>>>(Y, scl, sft, Xa, (long long)P1, (long long)P1, 64);
    // L2: 64 -> 128, then max over K
    dense_bn(Wl[2], Gl[2], Bl[2], 128, 64, 64, Xa, (int)P1, (int)P1, Y, Wh, scl, sft, stream);
    bn_relu_max_kernel<<<cdiv((long long)BATCH * 512 * 128, 256), 256, 0, stream>>>(Y, scl, sft, feats1, 512, 128);

    // ================= SA2 =================
    fps_kernel<<<BATCH, 256, 0, stream>>>(nx1, fidx2, 512, 128);
    gather_xyz_kernel<<<cdiv((long long)BATCH * 128 * 3, 256), 256, 0, stream>>>(nx1, fidx2, nx2, 512, 128);
    ball_query_kernel<<<cdiv((long long)BATCH * 128 * 32, 256), 256, 0, stream>>>(nx1, nx2, idx2, 512, 128, 0.4f * 0.4f);
    group_kernel<<<cdiv((long long)P2 * 160, 256), 256, 0, stream>>>(nx1, nx2, feats1, idx2, Xa, 512, 128, 128, 160);
    // L3: 131 -> 128
    dense_bn(Wl[3], Gl[3], Bl[3], 128, 131, 160, Xa, (int)P2, (int)P2, Y, Wh, scl, sft, stream);
    bn_relu_f16_kernel<<<cdiv((long long)P2 * 128, 256), 256, 0, stream>>>(Y, scl, sft, Xb, (long long)P2, (long long)P2, 128);
    // L4: 128 -> 128
    dense_bn(Wl[4], Gl[4], Bl[4], 128, 128, 128, Xb, (int)P2, (int)P2, Y, Wh, scl, sft, stream);
    bn_relu_f16_kernel<<<cdiv((long long)P2 * 128, 256), 256, 0, stream>>>(Y, scl, sft, Xa, (long long)P2, (long long)P2, 128);
    // L5: 128 -> 256, then max over K
    dense_bn(Wl[5], Gl[5], Bl[5], 256, 128, 128, Xa, (int)P2, (int)P2, Y, Wh, scl, sft, stream);
    bn_relu_max_kernel<<<cdiv((long long)BATCH * 128 * 256, 256), 256, 0, stream>>>(Y, scl, sft, feats2, 128, 256);

    // ================= SA3 (group-all) =================
    groupall_kernel<<<cdiv((long long)P3 * 288, 256), 256, 0, stream>>>(nx2, feats2, Xa, 128, 256, 288);
    // L6: 259 -> 256
    dense_bn(Wl[6], Gl[6], Bl[6], 256, 259, 288, Xa, (int)P3, (int)P3, Y, Wh, scl, sft, stream);
    bn_relu_f16_kernel<<<cdiv((long long)P3 * 256, 256), 256, 0, stream>>>(Y, scl, sft, Xb, (long long)P3, (long long)P3, 256);
    // L7: 256 -> 512
    dense_bn(Wl[7], Gl[7], Bl[7], 512, 256, 256, Xb, (int)P3, (int)P3, Y, Wh, scl, sft, stream);
    bn_relu_f16_kernel<<<cdiv((long long)P3 * 512, 256), 256, 0, stream>>>(Y, scl, sft, Xa, (long long)P3, (long long)P3, 512);
    // L8: 512 -> 1024, then max over S -> f16 [64][1024] (rows 16..63 zero)
    dense_bn(Wl[8], Gl[8], Bl[8], 1024, 512, 512, Xa, (int)P3, (int)P3, Y, Wh, scl, sft, stream);
    bn_relu_max_all_kernel<<<cdiv((long long)PFC * 1024, 256), 256, 0, stream>>>(Y, scl, sft, Xb, 128, 1024, PFC);

    // ================= FC head (rows padded to 64 for WMMA) =================
    // fc1: 1024 -> 512, BN over batch, relu
    dense_bn(fcW1, fcG1, fcB1, 512, 1024, 1024, Xb, PFC, BATCH, Y, Wh, scl, sft, stream);
    bn_relu_f16_kernel<<<cdiv((long long)PFC * 512, 256), 256, 0, stream>>>(Y, scl, sft, Xa, (long long)PFC, (long long)BATCH, 512);
    // fc2: 512 -> 256
    dense_bn(fcW2, fcG2, fcB2, 256, 512, 512, Xa, PFC, BATCH, Y, Wh, scl, sft, stream);
    bn_relu_f16_kernel<<<cdiv((long long)PFC * 256, 256), 256, 0, stream>>>(Y, scl, sft, Xb, (long long)PFC, (long long)BATCH, 256);
    // fc3: 256 -> 4096, sigmoid over the real 16 rows -> d_out
    wcvt_swz_kernel<<<cdiv((long long)4096 * 256, 256), 256, 0, stream>>>(fcW3, Wh, 4096, 256, 256);
    launch_gemm(Wh, Xb, Y, 4096, 256, PFC, stream);
    sigmoid_kernel<<<cdiv((long long)BATCH * 4096, 256), 256, 0, stream>>>(Y, (float*)d_out, (long long)BATCH * 4096);
}
